// MemoryRetriever_56607668961718
// MI455X (gfx1250) — matrix operand order repaired and checked by reading, not compiled, not run on hardware
//
#include <hip/hip_runtime.h>
#include <hip/hip_bf16.h>

typedef __bf16 bf16_t;
typedef __attribute__((ext_vector_type(16))) __bf16 v16bf;
typedef __attribute__((ext_vector_type(8)))  __bf16 v8bf;
typedef __attribute__((ext_vector_type(8)))  float  v8f;

#define Hh   1024
#define Ss   2048
#define BB   4
#define BS   8192      // B*S tokens
#define MEMM 8192
#define TOPK 2048
#define NHh  4
#define HDh  256

#define WMMA_BF16(a,b,c) \
  __builtin_amdgcn_wmma_f32_16x16x32_bf16(false,(a),false,(b),(short)0,(c),false,false)

// Load a v16bf A/B fragment as two contiguous 16-byte chunks.
__device__ inline v16bf ld2x8(const bf16_t* p0, const bf16_t* p1) {
  v8bf lo = *(const v8bf*)p0;
  v8bf hi = *(const v8bf*)p1;
  v16bf r;
#pragma unroll
  for (int i = 0; i < 8; i++) { r[i] = lo[i]; r[i + 8] = hi[i]; }
  return r;
}

__device__ inline float block_sum(float v, float* red) {
  int tid = threadIdx.x;
  red[tid] = v; __syncthreads();
  for (int s = 128; s > 0; s >>= 1) {
    if (tid < s) red[tid] += red[tid + s];
    __syncthreads();
  }
  float r = red[0]; __syncthreads();
  return r;
}

// ---------------- conversion kernels ----------------
__global__ void k_cvt(const float* __restrict__ in, bf16_t* __restrict__ out, int n) {
  int i = blockIdx.x * 256 + threadIdx.x;
  if (i < n) out[i] = (bf16_t)in[i];
}

__global__ void k_cvt_strided(const float* __restrict__ in, bf16_t* __restrict__ out,
                              int cols, int ldo, int n) {
  int i = blockIdx.x * 256 + threadIdx.x;
  if (i < n) out[(size_t)(i / cols) * ldo + (i % cols)] = (bf16_t)in[i];
}

// ---------------- top-k selection (exact jax.lax.top_k order) ----------------
__global__ __launch_bounds__(256) void k_topk(const float* __restrict__ imp,
                                              const float* __restrict__ rec,
                                              const float* __restrict__ acc,
                                              const float* __restrict__ selp,
                                              int* __restrict__ idx) {
  __shared__ float selv[MEMM];
  __shared__ float red[256];
  int tid = threadIdx.x;
  float mx = -1e30f;
  for (int j = tid; j < MEMM; j += 256) mx = fmaxf(mx, acc[j]);
  red[tid] = mx; __syncthreads();
  for (int s = 128; s > 0; s >>= 1) {
    if (tid < s) red[tid] = fmaxf(red[tid], red[tid + s]);
    __syncthreads();
  }
  float amax = red[0]; __syncthreads();
  float s0 = selp[0], s1 = selp[1], s2 = selp[2];
  float sm = fmaxf(s0, fmaxf(s1, s2));
  float e0 = __expf(s0 - sm), e1 = __expf(s1 - sm), e2 = __expf(s2 - sm);
  float den = e0 + e1 + e2;
  float w0 = e0 / den, w1 = e1 / den, w2 = e2 / den;
  for (int j = tid; j < MEMM; j += 256)
    selv[j] = w0 * imp[j] + w1 * rec[j] + w2 * (acc[j] / amax);
  __syncthreads();
  int i = blockIdx.x * 256 + tid;
  float si = selv[i];
  int rank = 0;
  for (int j = 0; j < MEMM; j++) {
    float sj = selv[j];
    rank += (sj > si) || (sj == si && j < i);
  }
  if (rank < TOPK) idx[rank] = i;
}

// ---------------- gather + LayerNorm -> bf16 ----------------
__global__ __launch_bounds__(256) void k_gather_ln(const float* __restrict__ mem,
                                                   const int* __restrict__ idx,
                                                   const float* __restrict__ g,
                                                   const float* __restrict__ bt,
                                                   bf16_t* __restrict__ out) {
  __shared__ float red[256];
  int row = blockIdx.x, tid = threadIdx.x;
  const float* src = mem + (size_t)idx[row] * Hh;
  float x[4]; float s = 0.f;
#pragma unroll
  for (int t = 0; t < 4; t++) { x[t] = src[t * 256 + tid]; s += x[t]; }
  s = block_sum(s, red);
  float mean = s * (1.f / Hh);
  float q = 0.f;
#pragma unroll
  for (int t = 0; t < 4; t++) { float d = x[t] - mean; q += d * d; }
  q = block_sum(q, red);
  float rstd = rsqrtf(q * (1.f / Hh) + 1e-5f);
#pragma unroll
  for (int t = 0; t < 4; t++) {
    int e = t * 256 + tid;
    out[(size_t)row * Hh + e] = (bf16_t)((x[t] - mean) * rstd * g[e] + bt[e]);
  }
}

// ---------------- WMMA GEMM: C = A(MxK) * W(NxK)^T + bias ----------------
template <typename OT, int ACT>
__device__ inline void epi(const v8f& a, OT* __restrict__ C, size_t ldc,
                           int row0, int n, float bv, int hi) {
#pragma unroll
  for (int r = 0; r < 8; r++) {
    float v = a[r] + bv;
    if (ACT == 1) v = 1.0f / (1.0f + __expf(-v));
    C[(size_t)(row0 + r + 8 * hi) * ldc + n] = (OT)v;
  }
}

template <typename OT, int ACT>
__global__ __launch_bounds__(256) void k_gemm(const bf16_t* __restrict__ A, int lda,
                                              const bf16_t* __restrict__ W, int K,
                                              const float* __restrict__ bias,
                                              OT* __restrict__ C, int ldc) {
  const int lane = threadIdx.x & 31, wave = threadIdx.x >> 5;
  const int hi = lane >> 4, ln = lane & 15, lo8 = hi * 8;
  const int m0 = blockIdx.y * 128 + (wave & 3) * 32;
  const int n0 = blockIdx.x * 64 + (wave >> 2) * 32;
  const bf16_t* a0p = A + (size_t)(m0 + ln) * lda;
  const bf16_t* a1p = a0p + (size_t)16 * lda;
  const bf16_t* b0p = W + (size_t)(n0 + ln) * K + hi * 8;
  const bf16_t* b1p = b0p + (size_t)16 * K;
  v8f c00 = {0.f,0.f,0.f,0.f,0.f,0.f,0.f,0.f};
  v8f c01 = c00, c10 = c00, c11 = c00;

  // software-pipelined main loop: load k-step (t+1) before WMMAs of k-step t
  v16bf a0 = ld2x8(a0p + lo8, a0p + 16 + lo8);
  v16bf a1 = ld2x8(a1p + lo8, a1p + 16 + lo8);
  v16bf b0 = ld2x8(b0p, b0p + 16);
  v16bf b1 = ld2x8(b1p, b1p + 16);
  for (int k0 = 0; k0 < K - 32; k0 += 32) {
    if (k0 + 256 < K) {   // warm L2/WGP$ ~8 k-steps ahead
      __builtin_prefetch(a0p + k0 + 256, 0, 3);
      __builtin_prefetch(b0p + k0 + 256, 0, 3);
    }
    int kn = k0 + 32;
    v16bf a0n = ld2x8(a0p + kn + lo8, a0p + kn + 16 + lo8);
    v16bf a1n = ld2x8(a1p + kn + lo8, a1p + kn + 16 + lo8);
    v16bf b0n = ld2x8(b0p + kn, b0p + kn + 16);
    v16bf b1n = ld2x8(b1p + kn, b1p + kn + 16);
    c00 = WMMA_BF16(a0, b0, c00);
    c01 = WMMA_BF16(a0, b1, c01);
    c10 = WMMA_BF16(a1, b0, c10);
    c11 = WMMA_BF16(a1, b1, c11);
    a0 = a0n; a1 = a1n; b0 = b0n; b1 = b1n;
  }
  c00 = WMMA_BF16(a0, b0, c00);
  c01 = WMMA_BF16(a0, b1, c01);
  c10 = WMMA_BF16(a1, b0, c10);
  c11 = WMMA_BF16(a1, b1, c11);

  float bv0 = bias[n0 + ln], bv1 = bias[n0 + 16 + ln];
  epi<OT, ACT>(c00, C, (size_t)ldc, m0,      n0 + ln,      bv0, hi);
  epi<OT, ACT>(c01, C, (size_t)ldc, m0,      n0 + 16 + ln, bv1, hi);
  epi<OT, ACT>(c10, C, (size_t)ldc, m0 + 16, n0 + ln,      bv0, hi);
  epi<OT, ACT>(c11, C, (size_t)ldc, m0 + 16, n0 + 16 + ln, bv1, hi);
}

// ---------------- bf16 transpose: Vt[d][kv] = V[kv][d] ----------------
__global__ void k_tr(const bf16_t* __restrict__ in, bf16_t* __restrict__ out) {
  __shared__ bf16_t t[32][33];
  int c0 = blockIdx.x * 32, r0 = blockIdx.y * 32;
  int x = threadIdx.x, y = threadIdx.y;
  for (int i = 0; i < 32; i += 8) t[y + i][x] = in[(size_t)(r0 + y + i) * Hh + c0 + x];
  __syncthreads();
  for (int i = 0; i < 32; i += 8) out[(size_t)(c0 + y + i) * TOPK + r0 + x] = t[x][y + i];
}

// ---------------- flash attention (online softmax, WMMA) ----------------
__global__ __launch_bounds__(256) void k_flash(const bf16_t* __restrict__ Q,
                                               const bf16_t* __restrict__ K,
                                               const bf16_t* __restrict__ Vt,
                                               bf16_t* __restrict__ ctx) {
  __shared__ bf16_t pst[8 * 512];
  const int bh = blockIdx.y, b = bh >> 2, h = bh & 3;
  const int wave = threadIdx.x >> 5, lane = threadIdx.x & 31;
  const int hi = lane >> 4, ln = lane & 15, lo8 = hi * 8;
  const int row0 = blockIdx.x * 128 + wave * 16;
  const bf16_t* qrow = Q + ((size_t)(b * Ss + row0 + ln)) * Hh + h * HDh;
  v16bf qf[8];
#pragma unroll
  for (int ks = 0; ks < 8; ks++)
    qf[ks] = ld2x8(qrow + ks * 32 + lo8, qrow + ks * 32 + 16 + lo8);
  v8f vz = {0.f,0.f,0.f,0.f,0.f,0.f,0.f,0.f};
  v8f o[16];
  float m8[8], l8[8];
#pragma unroll
  for (int t = 0; t < 16; t++) o[t] = vz;
#pragma unroll
  for (int r = 0; r < 8; r++) { m8[r] = -1e30f; l8[r] = 0.f; }
  bf16_t* sp = pst + wave * 512;
  const float SC = 0.0625f; // 1/sqrt(256)
  for (int kt = 0; kt < TOPK / 32; kt++) {
    const bf16_t* kb0 = K + ((size_t)(kt * 32 + ln)) * Hh + h * HDh + hi * 8;
    const bf16_t* kb1 = kb0 + (size_t)16 * Hh;
    v8f s0 = vz, s1 = vz;
    // double-buffered score loop
    v16bf bk0 = ld2x8(kb0, kb0 + 16);
    v16bf bk1 = ld2x8(kb1, kb1 + 16);
#pragma unroll
    for (int ks = 0; ks < 8; ks++) {
      v16bf nk0 = bk0, nk1 = bk1;
      if (ks < 7) {
        nk0 = ld2x8(kb0 + (ks + 1) * 32, kb0 + (ks + 1) * 32 + 16);
        nk1 = ld2x8(kb1 + (ks + 1) * 32, kb1 + (ks + 1) * 32 + 16);
      }
      s0 = WMMA_BF16(qf[ks], bk0, s0);
      s1 = WMMA_BF16(qf[ks], bk1, s1);
      bk0 = nk0; bk1 = nk1;
    }
    float c8[8];
#pragma unroll
    for (int r = 0; r < 8; r++) {
      float v = fmaxf(s0[r], s1[r]);
#pragma unroll
      for (int msk = 8; msk; msk >>= 1) v = fmaxf(v, __shfl_xor(v, msk, 32));
      float mn = fmaxf(m8[r], v * SC);
      float c = __expf(m8[r] - mn);
      m8[r] = mn; c8[r] = c; l8[r] *= c;
      float p0 = __expf(s0[r] * SC - mn);
      float p1 = __expf(s1[r] * SC - mn);
      float rs = p0 + p1;
#pragma unroll
      for (int msk = 8; msk; msk >>= 1) rs += __shfl_xor(rs, msk, 32);
      l8[r] += rs;
      sp[(r + 8 * hi) * 32 + ln]      = (bf16_t)p0;
      sp[(r + 8 * hi) * 32 + ln + 16] = (bf16_t)p1;
    }
#pragma unroll
    for (int t = 0; t < 16; t++)
#pragma unroll
      for (int r = 0; r < 8; r++) o[t][r] *= c8[r];
    asm volatile("s_wait_dscnt 0" ::: "memory"); // within-wave LDS RAW fence
    v16bf pa = ld2x8(sp + ln * 32 + lo8, sp + ln * 32 + 16 + lo8);
    const bf16_t* vb = Vt + ((size_t)(h * HDh + ln)) * TOPK + kt * 32 + hi * 8;
    // double-buffered PV loop
    v16bf bv = ld2x8(vb, vb + 16);
#pragma unroll
    for (int t = 0; t < 16; t++) {
      v16bf bvn = bv;
      if (t < 15)
        bvn = ld2x8(vb + (size_t)(t + 1) * 16 * TOPK,
                    vb + (size_t)(t + 1) * 16 * TOPK + 16);
      o[t] = WMMA_BF16(pa, bv, o[t]);
      bv = bvn;
    }
  }
#pragma unroll
  for (int t = 0; t < 16; t++)
#pragma unroll
    for (int r = 0; r < 8; r++) {
      float v = o[t][r] / l8[r];
      ctx[((size_t)(b * Ss + row0 + r + 8 * hi)) * Hh + h * HDh + t * 16 + ln] = (bf16_t)v;
    }
}

// ---------------- LN + exact GELU over 2048 cols -> bf16 ----------------
__global__ __launch_bounds__(256) void k_ln_gelu(const float* __restrict__ in,
                                                 const float* __restrict__ g,
                                                 const float* __restrict__ bt,
                                                 bf16_t* __restrict__ out) {
  __shared__ float red[256];
  int row = blockIdx.x, tid = threadIdx.x;
  const float* src = in + (size_t)row * 2048;
  float x[8]; float s = 0.f;
#pragma unroll
  for (int t = 0; t < 8; t++) { x[t] = src[t * 256 + tid]; s += x[t]; }
  s = block_sum(s, red);
  float mean = s * (1.f / 2048.f);
  float q = 0.f;
#pragma unroll
  for (int t = 0; t < 8; t++) { float d = x[t] - mean; q += d * d; }
  q = block_sum(q, red);
  float rstd = rsqrtf(q * (1.f / 2048.f) + 1e-5f);
#pragma unroll
  for (int t = 0; t < 8; t++) {
    int e = t * 256 + tid;
    float y = (x[t] - mean) * rstd * g[e] + bt[e];
    float ge = 0.5f * y * (1.0f + erff(y * 0.70710678118654752f));
    out[(size_t)row * 2048 + e] = (bf16_t)ge;
  }
}

// ---------------- final: LN(qh + gate*integ) -> f32 ----------------
__global__ __launch_bounds__(256) void k_final(const float* __restrict__ qh,
                                               const float* __restrict__ gate,
                                               const float* __restrict__ integ,
                                               const float* __restrict__ g,
                                               const float* __restrict__ bt,
                                               float* __restrict__ out) {
  __shared__ float red[256];
  int row = blockIdx.x, tid = threadIdx.x;
  size_t base = (size_t)row * Hh;
  float x[4]; float s = 0.f;
#pragma unroll
  for (int t = 0; t < 4; t++) {
    int e = t * 256 + tid;
    x[t] = qh[base + e] + gate[base + e] * integ[base + e];
    s += x[t];
  }
  s = block_sum(s, red);
  float mean = s * (1.f / Hh);
  float q = 0.f;
#pragma unroll
  for (int t = 0; t < 4; t++) { float d = x[t] - mean; q += d * d; }
  q = block_sum(q, red);
  float rstd = rsqrtf(q * (1.f / Hh) + 1e-5f);
#pragma unroll
  for (int t = 0; t < 4; t++) {
    int e = t * 256 + tid;
    out[base + e] = (x[t] - mean) * rstd * g[e] + bt[e];
  }
}

// ---------------- host launcher ----------------
extern "C" void kernel_launch(void* const* d_in, const int* in_sizes, int n_in,
                              void* d_out, int out_size, void* d_ws, size_t ws_size,
                              hipStream_t stream) {
  const float* qh     = (const float*)d_in[0];
  const float* memk   = (const float*)d_in[1];
  const float* imp    = (const float*)d_in[2];
  const float* rec    = (const float*)d_in[3];
  const float* accc   = (const float*)d_in[4];
  const float* Wq     = (const float*)d_in[5];
  const float* bq     = (const float*)d_in[6];
  const float* in_w   = (const float*)d_in[7];
  const float* in_b   = (const float*)d_in[8];
  const float* out_w  = (const float*)d_in[9];
  const float* out_b  = (const float*)d_in[10];
  const float* gate_w = (const float*)d_in[11];
  const float* gate_b = (const float*)d_in[12];
  const float* int_w1 = (const float*)d_in[13];
  const float* int_b1 = (const float*)d_in[14];
  const float* iln_g  = (const float*)d_in[15];
  const float* iln_b  = (const float*)d_in[16];
  const float* int_w2 = (const float*)d_in[17];
  const float* int_b2 = (const float*)d_in[18];
  const float* ln1_g  = (const float*)d_in[19];
  const float* ln1_b  = (const float*)d_in[20];
  const float* ln2_g  = (const float*)d_in[21];
  const float* ln2_b  = (const float*)d_in[22];
  const float* selp   = (const float*)d_in[23];

  const size_t MB = 1u << 20;
  size_t off = 0;
  auto alloc = [&](size_t bytes) -> void* {
    off = (off + 255) & ~(size_t)255;
    void* p = (char*)d_ws + off;
    off += bytes;
    return p;
  };
  int*    idx      = (int*)alloc((size_t)TOPK * 4);
  bf16_t* Wq_bf    = (bf16_t*)alloc((size_t)Hh * Hh * 2);
  bf16_t* inw_bf   = (bf16_t*)alloc((size_t)3 * Hh * Hh * 2);
  bf16_t* outw_bf  = (bf16_t*)alloc((size_t)Hh * Hh * 2);
  bf16_t* gatew_bf = (bf16_t*)alloc((size_t)Hh * 2 * Hh * 2);
  bf16_t* w1_bf    = (bf16_t*)alloc((size_t)2 * Hh * 2 * Hh * 2);
  bf16_t* w2_bf    = (bf16_t*)alloc((size_t)Hh * 2 * Hh * 2);
  bf16_t* cat      = (bf16_t*)alloc((size_t)BS * 2 * Hh * 2);   // 32MB; reused as integ f32
  char*   big      = (char*)alloc((size_t)64 * MB);             // reused as h1 f32
  bf16_t* q_bf   = (bf16_t*)(big);
  bf16_t* Q_bf   = (bf16_t*)(big + 16 * MB);
  bf16_t* memn   = (bf16_t*)(big + 32 * MB);
  bf16_t* K_bf   = (bf16_t*)(big + 36 * MB);
  bf16_t* V_bf   = (bf16_t*)(big + 40 * MB);
  bf16_t* Vt_bf  = (bf16_t*)(big + 44 * MB);
  bf16_t* ctx_bf = (bf16_t*)(big + 48 * MB);
  float*  h1_f   = (float*)big;                                 // reuse (64MB)
  bf16_t* h1g    = (bf16_t*)alloc((size_t)BS * 2 * Hh * 2);
  float*  gate_f = (float*)alloc((size_t)BS * Hh * 4);
  float*  integ_f = (float*)cat;                                // reuse (32MB)

  int n;
  n = Hh * Hh;         k_cvt<<<(n + 255) / 256, 256, 0, stream>>>(Wq, Wq_bf, n);
  n = 3 * Hh * Hh;     k_cvt<<<(n + 255) / 256, 256, 0, stream>>>(in_w, inw_bf, n);
  n = Hh * Hh;         k_cvt<<<(n + 255) / 256, 256, 0, stream>>>(out_w, outw_bf, n);
  n = Hh * 2 * Hh;     k_cvt<<<(n + 255) / 256, 256, 0, stream>>>(gate_w, gatew_bf, n);
  n = 2 * Hh * 2 * Hh; k_cvt<<<(n + 255) / 256, 256, 0, stream>>>(int_w1, w1_bf, n);
  n = Hh * 2 * Hh;     k_cvt<<<(n + 255) / 256, 256, 0, stream>>>(int_w2, w2_bf, n);
  n = BS * Hh;         k_cvt_strided<<<(n + 255) / 256, 256, 0, stream>>>(qh, cat, Hh, 2 * Hh, n);

  k_topk<<<MEMM / 256, 256, 0, stream>>>(imp, rec, accc, selp, idx);
  k_gather_ln<<<TOPK, 256, 0, stream>>>(memk, idx, ln1_g, ln1_b, memn);

  dim3 gBSxH(Hh / 64, BS / 128), gKVxH(Hh / 64, TOPK / 128), gBSx2H(2 * Hh / 64, BS / 128);
  // q = qh @ Wq^T + bq
  k_gemm<bf16_t, 0><<<gBSxH, 256, 0, stream>>>(cat, 2 * Hh, Wq_bf, Hh, bq, q_bf, Hh);
  // Q = q @ wq^T + bqi
  k_gemm<bf16_t, 0><<<gBSxH, 256, 0, stream>>>(q_bf, Hh, inw_bf, Hh, in_b, Q_bf, Hh);
  // K = mem_n @ wk^T + bki ; V = mem_n @ wv^T + bvi
  k_gemm<bf16_t, 0><<<gKVxH, 256, 0, stream>>>(memn, Hh, inw_bf + (size_t)Hh * Hh, Hh, in_b + Hh, K_bf, Hh);
  k_gemm<bf16_t, 0><<<gKVxH, 256, 0, stream>>>(memn, Hh, inw_bf + (size_t)2 * Hh * Hh, Hh, in_b + 2 * Hh, V_bf, Hh);
  k_tr<<<dim3(Hh / 32, TOPK / 32), dim3(32, 8), 0, stream>>>(V_bf, Vt_bf);
  // attention
  k_flash<<<dim3(Ss / 128, BB * NHh), 256, 0, stream>>>(Q_bf, K_bf, Vt_bf, ctx_bf);
  // attn_out -> second half of cat
  k_gemm<bf16_t, 0><<<gBSxH, 256, 0, stream>>>(ctx_bf, Hh, outw_bf, Hh, out_b, cat + Hh, 2 * Hh);
  // gate = sigmoid(cat @ gate_w^T + gate_b)
  k_gemm<float, 1><<<gBSxH, 256, 0, stream>>>(cat, 2 * Hh, gatew_bf, 2 * Hh, gate_b, gate_f, Hh);
  // h1 = cat @ int_w1^T + int_b1
  k_gemm<float, 0><<<gBSx2H, 256, 0, stream>>>(cat, 2 * Hh, w1_bf, 2 * Hh, int_b1, h1_f, 2 * Hh);
  // gelu(LN(h1))
  k_ln_gelu<<<BS, 256, 0, stream>>>(h1_f, iln_g, iln_b, h1g);
  // integ = h1g @ int_w2^T + int_b2
  k_gemm<float, 0><<<gBSxH, 256, 0, stream>>>(h1g, 2 * Hh, w2_bf, 2 * Hh, int_b2, integ_f, Hh);
  // out = LN(qh + gate*integ)
  k_final<<<BS, 256, 0, stream>>>(qh, gate_f, integ_f, ln2_g, ln2_b, (float*)d_out);
}
